// WTL_6133213299138
// MI455X (gfx1250) — matrix operand order repaired and checked by reading, not compiled, not run on hardware
//
#include <hip/hip_runtime.h>

typedef __attribute__((ext_vector_type(16))) _Float16 v16h;
typedef __attribute__((ext_vector_type(8)))  float    v8f;

namespace {

constexpr int NT = 256; // 8 waves of 32 (wave32)

struct Params {
  const float *x;
  const float *n1w, *n1b, *n2w, *n2b, *n3w, *n3b, *n4w, *n4b;
  const float *qkvw, *qkvb, *gaprojw, *gaprojb;
  const float *glqw, *glqb;            // provably dead (softmax over len-1 axis)
  const float *glkvw, *glkvb, *glprojw, *glprojb;
  const float *glsrw, *glsrb, *glnw, *glnb;
  const float *m1w1, *m1b1, *m1w2, *m1b2;
  const float *m2w1, *m2b1, *m2w2, *m2b2;
  float *out;
};

// A tile 16x32 f16 from row-major LDS (stride 32).
// Contiguous k order (same as loadBW): lane holds M=lane&15; VGPR v holds K = hs*16 + 2*v + {0,1}
__device__ __forceinline__ v16h loadA16(const _Float16 *base, int row0, int lane) {
  const _Float16 *rp = base + (row0 + (lane & 15)) * 32 + ((lane >> 4) & 1) * 16;
  v16h a;
#pragma unroll
  for (int v = 0; v < 8; ++v) {
    int k = v << 1;
    a[2 * v]     = rp[k];
    a[2 * v + 1] = rp[k + 1];
  }
  return a;
}

// B = W^T tile 32x16.  W is (out x 32) row-major f16 in LDS.
// B[k][n] = W[n][k]; lane n = lane&15 holds K = hs*16 + e  -> 16 contiguous halfs.
__device__ __forceinline__ v16h loadBW(const _Float16 *W, int n0, int lane) {
  const _Float16 *p = W + (n0 + (lane & 15)) * 32 + ((lane >> 4) & 1) * 16;
  v16h b;
#pragma unroll
  for (int e = 0; e < 16; ++e) b[e] = p[e];
  return b;
}

__device__ __forceinline__ float gelu_exact(float x) {
  return 0.5f * x * (1.0f + erff(x * 0.70710678118654752f));
}

__global__ __launch_bounds__(NT) void fused_block_kernel(Params P) {
  __shared__ float s_xres[2048];            // fp32 residual stream (2x32x32)
  __shared__ float s_f[6144];               // qkv f32 [64][96]; later LN3 f32 [64][32]
  __shared__ alignas(16) _Float16 s_a16[2048]; // LN outputs (WMMA A)
  __shared__ alignas(16) _Float16 s_o16[2048]; // attn-out / MLP hidden (WMMA A)
  __shared__ alignas(16) _Float16 s_wqkv[3072];
  __shared__ alignas(16) _Float16 s_wga[1024];
  __shared__ alignas(16) _Float16 s_w11[1024];
  __shared__ alignas(16) _Float16 s_w12[1024];
  __shared__ alignas(16) _Float16 s_w21[1024];
  __shared__ alignas(16) _Float16 s_w22[1024];
  __shared__ float s_ln[8 * 32];
  __shared__ float s_qkvb[96];
  __shared__ float s_gab[32];
  __shared__ float s_b11[32], s_b12[32], s_b21[32], s_b22[32];
  __shared__ float s_glpb[32];
  __shared__ float s_P4[128];               // gl_proj folded mod-4: [32][4]
  __shared__ float s_kvw[32], s_kvb[8];
  __shared__ float s_srw[256], s_srb[2], s_nw4[4], s_nb4[4];
  __shared__ float s_conv[16], s_v[32];

  const int tid  = threadIdx.x;
  const int lane = tid & 31;
  const int wv   = tid >> 5;
  const int b    = blockIdx.x;

  auto cp32 = [&](float *d, const float *s, int n) {
    for (int i = tid; i < n; i += NT) d[i] = s[i];
  };
  auto cp16 = [&](_Float16 *d, const float *s, int n) {
    for (int i = tid; i < n; i += NT) d[i] = (_Float16)s[i];
  };
  // LN over feature axis (32) -> f16 A operand
  auto layer_norm16 = [&](const float *w, const float *bb) {
    if (tid < 64) {
      const float *xr = &s_xres[tid * 32];
      float m = 0.f;
#pragma unroll
      for (int j = 0; j < 32; ++j) m += xr[j];
      m *= 0.03125f;
      float vs = 0.f;
#pragma unroll
      for (int j = 0; j < 32; ++j) { float d = xr[j] - m; vs += d * d; }
      float inv = rsqrtf(vs * 0.03125f + 1e-6f);
#pragma unroll
      for (int j = 0; j < 32; ++j)
        s_a16[tid * 32 + j] = (_Float16)((xr[j] - m) * inv * w[j] + bb[j]);
    }
  };
  // one 64x32 @ 32x32 matmul (8 tiles, one per wave), += into residual
  auto mm_residual = [&](const _Float16 *A, const _Float16 *W, const float *bias) {
    int mt = wv >> 1, nt = wv & 1;
    v16h a  = loadA16(A, mt * 16, lane);
    v16h bw = loadBW(W, nt * 16, lane);
    v8f acc = {};
    acc = __builtin_amdgcn_wmma_f32_16x16x32_f16(false, a, false, bw, (short)0,
                                                 acc, false, false);
    int col  = nt * 16 + (lane & 15);
    int row0 = mt * 16 + ((lane >> 4) & 1) * 8;
    float bb = bias[col];
#pragma unroll
    for (int r = 0; r < 8; ++r) s_xres[(row0 + r) * 32 + col] += acc[r] + bb;
  };
  // one 64x32 @ 32x32 matmul, GELU, -> f16 hidden buffer
  auto mm_gelu = [&](const _Float16 *A, const _Float16 *W, const float *bias) {
    int mt = wv >> 1, nt = wv & 1;
    v16h a  = loadA16(A, mt * 16, lane);
    v16h bw = loadBW(W, nt * 16, lane);
    v8f acc = {};
    acc = __builtin_amdgcn_wmma_f32_16x16x32_f16(false, a, false, bw, (short)0,
                                                 acc, false, false);
    int col  = nt * 16 + (lane & 15);
    int row0 = mt * 16 + ((lane >> 4) & 1) * 8;
    float bb = bias[col];
#pragma unroll
    for (int r = 0; r < 8; ++r)
      s_o16[(row0 + r) * 32 + col] = (_Float16)gelu_exact(acc[r] + bb);
  };

  // ---------- stage weights (f16) / params, load x slice ----------
  cp16(s_wqkv, P.qkvw, 3072);
  cp16(s_wga,  P.gaprojw, 1024);
  cp16(s_w11,  P.m1w1, 1024);  cp16(s_w12, P.m1w2, 1024);
  cp16(s_w21,  P.m2w1, 1024);  cp16(s_w22, P.m2w2, 1024);
  cp32(s_ln +   0, P.n1w, 32); cp32(s_ln +  32, P.n1b, 32);
  cp32(s_ln +  64, P.n2w, 32); cp32(s_ln +  96, P.n2b, 32);
  cp32(s_ln + 128, P.n3w, 32); cp32(s_ln + 160, P.n3b, 32);
  cp32(s_ln + 192, P.n4w, 32); cp32(s_ln + 224, P.n4b, 32);
  cp32(s_qkvb, P.qkvb, 96);  cp32(s_gab, P.gaprojb, 32);
  cp32(s_b11, P.m1b1, 32);   cp32(s_b12, P.m1b2, 32);
  cp32(s_b21, P.m2b1, 32);   cp32(s_b22, P.m2b2, 32);
  cp32(s_glpb, P.glprojb, 32);
  cp32(s_kvw, P.glkvw, 32);  cp32(s_kvb, P.glkvb, 8);
  cp32(s_srw, P.glsrw, 256); cp32(s_srb, P.glsrb, 2);
  cp32(s_nw4, P.glnw, 4);    cp32(s_nb4, P.glnb, 4);
  // P4[j][d] = sum_{w%4==d} gl_proj_w[j][w]  (global-attn rank-4 collapse)
  for (int i = tid; i < 128; i += NT) {
    int j = i >> 2, d = i & 3;
    float s = 0.f;
    for (int w = d; w < 32; w += 4) s += P.glprojw[j * 32 + w];
    s_P4[i] = s;
  }
  {
    const float4 *g4 = (const float4 *)(P.x + (size_t)b * 2048);
    float4 *r4 = (float4 *)s_xres;
    for (int i = tid; i < 512; i += NT) r4[i] = g4[i];
  }
  __syncthreads();

  // ---------- sublayer 1: group attention ----------
  layer_norm16(s_ln + 0, s_ln + 32);
  __syncthreads();

  // qkv = LN1(x)(64x32) @ Wqkv^T(32x96) + b : 24 WMMA tiles
  for (int t = wv; t < 24; t += 8) {
    int mt = t / 6, nt = t % 6;
    v16h a  = loadA16(s_a16, mt * 16, lane);
    v16h bw = loadBW(s_wqkv, nt * 16, lane);
    v8f acc = {};
    acc = __builtin_amdgcn_wmma_f32_16x16x32_f16(false, a, false, bw, (short)0,
                                                 acc, false, false);
    int col  = nt * 16 + (lane & 15);
    int row0 = mt * 16 + ((lane >> 4) & 1) * 8;
    float bb = s_qkvb[col];
#pragma unroll
    for (int r = 0; r < 8; ++r) s_f[(row0 + r) * 96 + col] = acc[r] + bb;
  }
  __syncthreads();

  // window attention: unit = (c, g, head); tokens live inside 96 feats of 2 rows
  if (tid < 128) {
    int c = tid >> 6, g = (tid >> 2) & 15, n = tid & 3;
    int h0 = (g >> 2) * 8 + (g & 3) * 2;
    float q0[8], q1[8], k0[8], k1[8], v0[8], v1[8];
#pragma unroll
    for (int t = 0; t < 8; ++t) {
      const float *p = &s_f[(c * 32 + h0 + (t >> 2)) * 96 + (t & 3) * 24 + n * 2];
      q0[t] = p[0];  q1[t] = p[1];
      k0[t] = p[8];  k1[t] = p[9];
      v0[t] = p[16]; v1[t] = p[17];
    }
    const float scale = 0.35355339059327373f; // (IMG/NH)^-0.5 = 8^-0.5
#pragma unroll
    for (int i = 0; i < 8; ++i) {
      float l[8], mx = -3.4e38f;
#pragma unroll
      for (int j = 0; j < 8; ++j) {
        l[j] = scale * (q0[i] * k0[j] + q1[i] * k1[j]);
        mx = fmaxf(mx, l[j]);
      }
      float s = 0.f, o0 = 0.f, o1 = 0.f;
#pragma unroll
      for (int j = 0; j < 8; ++j) {
        float e = __expf(l[j] - mx);
        s += e; o0 += e * v0[j]; o1 += e * v1[j];
      }
      float rinv = 1.f / s;
      o0 *= rinv; o1 *= rinv;
      // scatter via the reference's reshape/swap permutation
      int flat = c * 1024 + n * 256 + g * 16 + i * 2;
#pragma unroll
      for (int d = 0; d < 2; ++d) {
        int f  = flat + d;
        int a1 = f >> 9, w1 = (f >> 7) & 3, s1 = (f >> 4) & 7,
            s2 = (f >> 1) & 7, cc = f & 1;
        int f2 = a1 * 512 + s1 * 64 + w1 * 16 + s2 * 2 + cc;
        s_o16[f2] = (_Float16)(d ? o1 : o0);
      }
    }
  }
  __syncthreads();

  mm_residual(s_o16, s_wga, s_gab);          // group proj + residual
  __syncthreads();

  // ---------- sublayer 2: MLP1 ----------
  layer_norm16(s_ln + 64, s_ln + 96);
  __syncthreads();
  mm_gelu(s_a16, s_w11, s_b11);
  __syncthreads();
  mm_residual(s_o16, s_w12, s_b12);
  __syncthreads();

  // ---------- sublayer 3: global attention (softmax over len-1 => v broadcast) ----------
  if (tid < 64) {  // LN3 in f32 (feeds conv)
    const float *xr = &s_xres[tid * 32];
    float m = 0.f;
#pragma unroll
    for (int j = 0; j < 32; ++j) m += xr[j];
    m *= 0.03125f;
    float vs = 0.f;
#pragma unroll
    for (int j = 0; j < 32; ++j) { float d = xr[j] - m; vs += d * d; }
    float inv = rsqrtf(vs * 0.03125f + 1e-6f);
    const float *w = s_ln + 128, *bb = s_ln + 160;
#pragma unroll
    for (int j = 0; j < 32; ++j) s_f[tid * 32 + j] = (xr[j] - m) * inv * w[j] + bb[j];
  }
  __syncthreads();

  if (tid < 16) { // strided conv: only output rows y=2,3 reach v
    int oc = tid >> 3, yy = 2 + ((tid >> 2) & 1), xx = tid & 3;
    float s = s_srb[oc];
#pragma unroll
    for (int ic = 0; ic < 2; ++ic) {
      const float *wp = &s_srw[(oc * 2 + ic) * 64];
      const float *ip = &s_f[(ic * 32 + yy * 8) * 32 + xx * 8];
#pragma unroll
      for (int ky = 0; ky < 8; ++ky)
#pragma unroll
        for (int kx = 0; kx < 8; ++kx) s += ip[ky * 32 + kx] * wp[ky * 8 + kx];
    }
    s_conv[(oc * 2 + (yy - 2)) * 4 + xx] = s;
  }
  __syncthreads();

  if (tid < 4) { // LN(4) + kv linear -> v values
    int c = tid >> 1, r = 2 + (tid & 1);
    const float *cs = &s_conv[(c * 2 + (r - 2)) * 4];
    float m = 0.25f * (cs[0] + cs[1] + cs[2] + cs[3]);
    float vs = 0.f;
#pragma unroll
    for (int d = 0; d < 4; ++d) { float e = cs[d] - m; vs += e * e; }
    float inv = rsqrtf(vs * 0.25f + 1e-5f);
    float xs[4];
#pragma unroll
    for (int d = 0; d < 4; ++d) xs[d] = (cs[d] - m) * inv * s_nw4[d] + s_nb4[d];
#pragma unroll
    for (int fo = 0; fo < 8; ++fo) {
      float kv = s_kvb[fo];
#pragma unroll
      for (int dc = 0; dc < 4; ++dc) kv += xs[dc] * s_kvw[fo * 4 + dc];
      int d1 = (r & 1) * 2 + (fo >> 2), d2 = fo & 3;
      s_v[c * 16 + d1 * 4 + d2] = kv;
    }
  }
  __syncthreads();

  // rank-4 projection of the broadcast v, + residual
  for (int idx = tid; idx < 2048; idx += NT) {
    int r2 = idx >> 5, j = idx & 31;
    int c2 = r2 >> 5, h2 = r2 & 31;
    int h = c2 * 2 + (h2 >> 4);
    int c = (h2 >> 3) & 1;
    const float *vv = &s_v[c * 16 + h * 4];
    float s = s_glpb[j];
#pragma unroll
    for (int dd = 0; dd < 4; ++dd) s += vv[dd] * s_P4[j * 4 + dd];
    s_xres[idx] += s;
  }
  __syncthreads();

  // ---------- sublayer 4: MLP2 ----------
  layer_norm16(s_ln + 192, s_ln + 224);
  __syncthreads();
  mm_gelu(s_a16, s_w21, s_b21);
  __syncthreads();
  mm_residual(s_o16, s_w22, s_b22);
  __syncthreads();

  // ---------- store ----------
  {
    float4 *o4 = (float4 *)(P.out + (size_t)b * 2048);
    const float4 *r4 = (const float4 *)s_xres;
    for (int i = tid; i < 512; i += NT) o4[i] = r4[i];
  }
}

} // namespace

extern "C" void kernel_launch(void *const *d_in, const int *in_sizes, int n_in,
                              void *d_out, int out_size, void *d_ws, size_t ws_size,
                              hipStream_t stream) {
  (void)n_in; (void)out_size; (void)d_ws; (void)ws_size;
  Params P;
  P.x       = (const float *)d_in[0];
  P.n1w     = (const float *)d_in[1];  P.n1b   = (const float *)d_in[2];
  P.n2w     = (const float *)d_in[3];  P.n2b   = (const float *)d_in[4];
  P.n3w     = (const float *)d_in[5];  P.n3b   = (const float *)d_in[6];
  P.n4w     = (const float *)d_in[7];  P.n4b   = (const float *)d_in[8];
  P.qkvw    = (const float *)d_in[9];  P.qkvb  = (const float *)d_in[10];
  P.gaprojw = (const float *)d_in[11]; P.gaprojb = (const float *)d_in[12];
  P.glqw    = (const float *)d_in[13]; P.glqb  = (const float *)d_in[14];
  P.glkvw   = (const float *)d_in[15]; P.glkvb = (const float *)d_in[16];
  P.glprojw = (const float *)d_in[17]; P.glprojb = (const float *)d_in[18];
  P.glsrw   = (const float *)d_in[19]; P.glsrb = (const float *)d_in[20];
  P.glnw    = (const float *)d_in[21]; P.glnb  = (const float *)d_in[22];
  P.m1w1    = (const float *)d_in[23]; P.m1b1  = (const float *)d_in[24];
  P.m1w2    = (const float *)d_in[25]; P.m1b2  = (const float *)d_in[26];
  P.m2w1    = (const float *)d_in[27]; P.m2b1  = (const float *)d_in[28];
  P.m2w2    = (const float *)d_in[29]; P.m2b2  = (const float *)d_in[30];
  P.out     = (float *)d_out;

  int B = in_sizes[0] / 2048; // (8192) images of 2x32x32
  fused_block_kernel<<<B, NT, 0, stream>>>(P);
}
